// CrossAttentionLayer_88940182766344
// MI455X (gfx1250) — matrix operand order repaired and checked by reading, not compiled, not run on hardware
//
#include <hip/hip_runtime.h>
#include <hip/hip_bf16.h>

typedef _Float16 f16;
typedef __attribute__((ext_vector_type(16))) _Float16 v16h;
typedef __attribute__((ext_vector_type(8)))  _Float16 v8h;
typedef __attribute__((ext_vector_type(4)))  _Float16 v4h;
typedef __attribute__((ext_vector_type(2)))  __fp16   v2hf;   // return type of cvt_pkrtz
typedef __attribute__((ext_vector_type(8)))  float    v8f;

#define EMBED 256
#define NHEAD 4
#define HDIM  64
#define SEQ   2048
#define BATCH 4
#define MTOT  (BATCH*SEQ)   /* 8192 rows */

static __device__ __forceinline__ v8f wmma_f16(v16h a, v16h b, v8f c) {
    return __builtin_amdgcn_wmma_f32_16x16x32_f16(
        false, a, false, b, (short)0, c, false, false);
}

// packed f32x4 -> f16x4 (v_cvt_pk_rtz_f16_f32 x2)
static __device__ __forceinline__ v4h cvt4(float x, float y, float z, float w) {
    v2hf a = __builtin_amdgcn_cvt_pkrtz(x, y);
    v2hf b = __builtin_amdgcn_cvt_pkrtz(z, w);
    v4h r;
    r.x = (f16)a.x; r.y = (f16)a.y; r.z = (f16)b.x; r.w = (f16)b.y;
    return r;
}

// A-fragment (16x32 f16) for lane (m = lane&15, h = lane>>4):
//   k = e + 16h (+32kc)
static __device__ __forceinline__ v16h load_afrag(const f16* __restrict__ row, int h, int kc) {
    v16h a;
    const v8h lo = *(const v8h*)(row + 16*h + 32*kc);
    const v8h hi = *(const v8h*)(row + 8 + 16*h + 32*kc);
    f16* ap = (f16*)&a;
    const f16* lp = (const f16*)&lo;
    const f16* hp = (const f16*)&hi;
#pragma unroll
    for (int i = 0; i < 8; ++i) { ap[i] = lp[i]; ap[8 + i] = hp[i]; }
    return a;
}

// ---- CDNA5 async global->LDS (16B per lane), tracked by ASYNCcnt ----------
static __device__ __forceinline__ void async_b128(unsigned int lds_byte, const void* g) {
    asm volatile("global_load_async_to_lds_b128 %0, %1, off"
                 :: "v"(lds_byte), "v"((unsigned long long)(uintptr_t)g)
                 : "memory");
}
static __device__ __forceinline__ void wait_async_le4() {
    asm volatile("s_wait_asynccnt 0x4" ::: "memory");
}
static __device__ __forceinline__ void wait_async_0() {
    asm volatile("s_wait_asynccnt 0x0" ::: "memory");
}

// ---------------------------------------------------------------------------
// Projection GEMM  (M x 256) @ (256 x Ncols) + bias, f32 in, f16 out.
// Q path (Ncols=256): dst0 = Q [B,H,N,D].
// KV path (Ncols=512): dst0 = K [B,H,N,D], dst1 = V TRANSPOSED [B,H,D,N].
// Block = 128 thr (4 waves), tile = 128 rows x 64 cols, K-tiles of 64.
// ---------------------------------------------------------------------------
__global__ __launch_bounds__(128, 1)
void proj_kernel(const float* __restrict__ X, const float* __restrict__ W,
                 const float* __restrict__ bias,
                 f16* __restrict__ dst0, f16* __restrict__ dst1, int Ncols)
{
    __shared__ __align__(32) f16 ldsA [128][80];  // rows x k (row 160B, 32B-aligned)
    __shared__ __align__(32) f16 ldsWt[64][80];   // cols x k (transposed W)

    const int tid  = threadIdx.x;
    const int wave = tid >> 5, lane = tid & 31;
    const int lm = lane & 15, lh = lane >> 4;
    const int rowBase = blockIdx.x * 128;
    const int colBase = blockIdx.y * 64;

    v8f acc[2][4] = {};

    for (int kt = 0; kt < 4; ++kt) {
        __syncthreads();
        // A: 128 rows x 64 k, float4 loads -> packed cvt -> b64 LDS stores
#pragma unroll
        for (int i = 0; i < 16; ++i) {
            int idx = tid + i * 128;            // 0..2047 float4 chunks
            int r = idx >> 4, c4 = idx & 15;
            const float4 w = *(const float4*)(X + (size_t)(rowBase + r) * EMBED + kt * 64 + c4 * 4);
            *(v4h*)(&ldsA[r][c4 * 4]) = cvt4(w.x, w.y, w.z, w.w);
        }
        // W: 64 k x 64 n, register-blocked 4x4 transpose (vector both sides)
#pragma unroll
        for (int i = 0; i < 2; ++i) {
            int bb = tid + i * 128;             // 0..255 4x4 blocks
            int k0 = (bb >> 4) * 4, n0 = (bb & 15) * 4;
            float4 w0 = *(const float4*)(W + (size_t)(kt * 64 + k0 + 0) * Ncols + colBase + n0);
            float4 w1 = *(const float4*)(W + (size_t)(kt * 64 + k0 + 1) * Ncols + colBase + n0);
            float4 w2 = *(const float4*)(W + (size_t)(kt * 64 + k0 + 2) * Ncols + colBase + n0);
            float4 w3 = *(const float4*)(W + (size_t)(kt * 64 + k0 + 3) * Ncols + colBase + n0);
            *(v4h*)(&ldsWt[n0 + 0][k0]) = cvt4(w0.x, w1.x, w2.x, w3.x);
            *(v4h*)(&ldsWt[n0 + 1][k0]) = cvt4(w0.y, w1.y, w2.y, w3.y);
            *(v4h*)(&ldsWt[n0 + 2][k0]) = cvt4(w0.z, w1.z, w2.z, w3.z);
            *(v4h*)(&ldsWt[n0 + 3][k0]) = cvt4(w0.w, w1.w, w2.w, w3.w);
        }
        __syncthreads();
#pragma unroll
        for (int kc = 0; kc < 2; ++kc) {
            v16h a0 = load_afrag(&ldsA[wave * 16 + lm][0],      lh, kc);
            v16h a1 = load_afrag(&ldsA[64 + wave * 16 + lm][0], lh, kc);
#pragma unroll
            for (int nc = 0; nc < 4; ++nc) {
                v16h b = *(const v16h*)(&ldsWt[nc * 16 + lm][16 * lh + 32 * kc]);
                acc[0][nc] = wmma_f16(a0, b, acc[0][nc]);
                acc[1][nc] = wmma_f16(a1, b, acc[1][nc]);
            }
        }
    }

    // Epilogue: +bias, route (Q/K head-major; V transposed d-major).
#pragma unroll
    for (int mt = 0; mt < 2; ++mt) {
#pragma unroll
        for (int nc = 0; nc < 4; ++nc) {
            const float* av = (const float*)&acc[mt][nc];
#pragma unroll
            for (int r = 0; r < 8; ++r) {
                int m  = r + 8 * lh;
                int gr = rowBase + mt * 64 + wave * 16 + m;
                int c  = colBase + nc * 16 + lm;
                float v = av[r] + bias[c];
                int b  = gr >> 11, nq = gr & (SEQ - 1);
                if (c < EMBED) {
                    int head = c >> 6, d = c & (HDIM - 1);
                    dst0[(((size_t)(b * NHEAD + head)) * SEQ + nq) * HDIM + d] = (f16)v;
                } else {
                    int cc = c - EMBED;
                    int head = cc >> 6, d = cc & (HDIM - 1);
                    dst1[(((size_t)(b * NHEAD + head)) * HDIM + d) * SEQ + nq] = (f16)v;
                }
            }
        }
    }
}

// ---------------------------------------------------------------------------
// Flash attention.  Grid = (SEQ/64, B*H), block = 128 (4 waves).
// K/V tiles double-buffered via global_load_async_to_lds_b128.
// ---------------------------------------------------------------------------
__global__ __launch_bounds__(128, 1)
void flash_kernel(const f16* __restrict__ Qw, const f16* __restrict__ Kw,
                  const f16* __restrict__ Vt, float* __restrict__ O)
{
    __shared__ __align__(32) f16 ldsK [2][32][64];  // kv x d, row-major (4KB/buf)
    __shared__ __align__(32) f16 ldsVt[2][64][32];  // d x kv              (4KB/buf)
    __shared__ __align__(32) f16 ldsP [4][16][32];  // per-wave P tile

    const int tid  = threadIdx.x;
    const int wave = tid >> 5, lane = tid & 31;
    const int lm = lane & 15, lh = lane >> 4;
    const int bh    = blockIdx.y;
    const int qBase = blockIdx.x * 64;

    const f16* Kb  = Kw + (size_t)bh * SEQ * HDIM;   // [kv][d]
    const f16* Vtb = Vt + (size_t)bh * HDIM * SEQ;   // [d][kv]
    const unsigned int kLds = (unsigned int)(uintptr_t)&ldsK[0][0][0];
    const unsigned int vLds = (unsigned int)(uintptr_t)&ldsVt[0][0][0];

    // Q A-fragments, pre-scaled by SCALE * log2(e).
    const float qscale = 0.125f * 1.4426950408889634f;
    v16h qa[2];
    {
        const f16* qrow = Qw + (size_t)bh * SEQ * HDIM + (size_t)(qBase + wave * 16 + lm) * HDIM;
#pragma unroll
        for (int kc = 0; kc < 2; ++kc) {
            v16h a = load_afrag(qrow, lh, kc);
            f16* ap = (f16*)&a;
#pragma unroll
            for (int i = 0; i < 16; ++i) ap[i] = (f16)((float)ap[i] * qscale);
            qa[kc] = a;
        }
    }

    float mrun[8], lrun[8];
#pragma unroll
    for (int r = 0; r < 8; ++r) { mrun[r] = -1e30f; lrun[r] = 0.0f; }
    v8f o[4] = {};

    // Issue one KV tile (4 async b128 per thread: 2 K chunks + 2 V chunks).
    auto issue_tile = [&](int t, int buf) {
#pragma unroll
        for (int j = 0; j < 2; ++j) {
            int c = tid * 2 + j;                       // 0..255 16B chunks
            async_b128(kLds + buf * 4096 + c * 16, Kb + (size_t)t * 32 * HDIM + c * 8);
            int d = c >> 2, kv0 = (c & 3) * 8;
            async_b128(vLds + buf * 4096 + c * 16, Vtb + (size_t)d * SEQ + t * 32 + kv0);
        }
    };

    const int NT = SEQ / 32;
    issue_tile(0, 0);

    for (int t = 0; t < NT; ++t) {
        const int buf = t & 1;
        if (t + 1 < NT) { issue_tile(t + 1, buf ^ 1); wait_async_le4(); }
        else            { wait_async_0(); }
        __syncthreads();

        // S = Q * K^T : 4 WMMAs
        v8f sacc[2] = {};
#pragma unroll
        for (int kc = 0; kc < 2; ++kc) {
#pragma unroll
            for (int nc = 0; nc < 2; ++nc) {
                v16h b = *(const v16h*)(&ldsK[buf][nc * 16 + lm][16 * lh + 32 * kc]);
                sacc[nc] = wmma_f16(qa[kc], b, sacc[nc]);
            }
        }
        float* s0 = (float*)&sacc[0];
        float* s1 = (float*)&sacc[1];

        // Online softmax (row m = r + 8*lh; reductions within 16-lane half)
#pragma unroll
        for (int r = 0; r < 8; ++r) {
            float tm = fmaxf(s0[r], s1[r]);
#pragma unroll
            for (int off = 1; off < 16; off <<= 1) tm = fmaxf(tm, __shfl_xor(tm, off, 32));
            float mn = fmaxf(mrun[r], tm);
            float alpha = exp2f(mrun[r] - mn);
            float p0 = exp2f(s0[r] - mn);
            float p1 = exp2f(s1[r] - mn);
            float ts = p0 + p1;
#pragma unroll
            for (int off = 1; off < 16; off <<= 1) ts += __shfl_xor(ts, off, 32);
            lrun[r] = lrun[r] * alpha + ts;
            mrun[r] = mn;
#pragma unroll
            for (int nc = 0; nc < 4; ++nc) ((float*)&o[nc])[r] *= alpha;
            int m = r + 8 * lh;
            ldsP[wave][m][lm]      = (f16)p0;
            ldsP[wave][m][16 + lm] = (f16)p1;
        }

        // O += P * V : 4 WMMAs
        v16h pa = load_afrag(&ldsP[wave][lm][0], lh, 0);
#pragma unroll
        for (int nc = 0; nc < 4; ++nc) {
            v16h vb = *(const v16h*)(&ldsVt[buf][nc * 16 + lm][16 * lh]);
            o[nc] = wmma_f16(pa, vb, o[nc]);
        }
        __syncthreads();   // protect buf^1 before its refill next iteration
    }

    // Normalize, store f32 attention output in [B, N, C].
    const int b = bh >> 2, head = bh & 3;
    float inv[8];
#pragma unroll
    for (int r = 0; r < 8; ++r) inv[r] = 1.0f / lrun[r];
#pragma unroll
    for (int nc = 0; nc < 4; ++nc) {
        const float* ov = (const float*)&o[nc];
#pragma unroll
        for (int r = 0; r < 8; ++r) {
            int m  = r + 8 * lh;
            int nq = qBase + wave * 16 + m;
            int c  = head * HDIM + nc * 16 + lm;
            O[((size_t)b * SEQ + nq) * EMBED + c] = ov[r] * inv[r];
        }
    }
}

// ---------------------------------------------------------------------------
// Output projection + bias + residual, f32 in/out. 128x64 tiles like proj.
// ---------------------------------------------------------------------------
__global__ __launch_bounds__(128, 1)
void oproj_kernel(const float* __restrict__ X, const float* __restrict__ W,
                  const float* __restrict__ bias, const float* __restrict__ resid,
                  float* __restrict__ Y)
{
    __shared__ __align__(32) f16 ldsA [128][80];
    __shared__ __align__(32) f16 ldsWt[64][80];

    const int tid  = threadIdx.x;
    const int wave = tid >> 5, lane = tid & 31;
    const int lm = lane & 15, lh = lane >> 4;
    const int rowBase = blockIdx.x * 128;
    const int colBase = blockIdx.y * 64;

    v8f acc[2][4] = {};

    for (int kt = 0; kt < 4; ++kt) {
        __syncthreads();
#pragma unroll
        for (int i = 0; i < 16; ++i) {
            int idx = tid + i * 128;
            int r = idx >> 4, c4 = idx & 15;
            const float4 w = *(const float4*)(X + (size_t)(rowBase + r) * EMBED + kt * 64 + c4 * 4);
            *(v4h*)(&ldsA[r][c4 * 4]) = cvt4(w.x, w.y, w.z, w.w);
        }
#pragma unroll
        for (int i = 0; i < 2; ++i) {
            int bb = tid + i * 128;
            int k0 = (bb >> 4) * 4, n0 = (bb & 15) * 4;
            float4 w0 = *(const float4*)(W + (size_t)(kt * 64 + k0 + 0) * EMBED + colBase + n0);
            float4 w1 = *(const float4*)(W + (size_t)(kt * 64 + k0 + 1) * EMBED + colBase + n0);
            float4 w2 = *(const float4*)(W + (size_t)(kt * 64 + k0 + 2) * EMBED + colBase + n0);
            float4 w3 = *(const float4*)(W + (size_t)(kt * 64 + k0 + 3) * EMBED + colBase + n0);
            *(v4h*)(&ldsWt[n0 + 0][k0]) = cvt4(w0.x, w1.x, w2.x, w3.x);
            *(v4h*)(&ldsWt[n0 + 1][k0]) = cvt4(w0.y, w1.y, w2.y, w3.y);
            *(v4h*)(&ldsWt[n0 + 2][k0]) = cvt4(w0.z, w1.z, w2.z, w3.z);
            *(v4h*)(&ldsWt[n0 + 3][k0]) = cvt4(w0.w, w1.w, w2.w, w3.w);
        }
        __syncthreads();
#pragma unroll
        for (int kc = 0; kc < 2; ++kc) {
            v16h a0 = load_afrag(&ldsA[wave * 16 + lm][0],      lh, kc);
            v16h a1 = load_afrag(&ldsA[64 + wave * 16 + lm][0], lh, kc);
#pragma unroll
            for (int nc = 0; nc < 4; ++nc) {
                v16h b = *(const v16h*)(&ldsWt[nc * 16 + lm][16 * lh + 32 * kc]);
                acc[0][nc] = wmma_f16(a0, b, acc[0][nc]);
                acc[1][nc] = wmma_f16(a1, b, acc[1][nc]);
            }
        }
    }

#pragma unroll
    for (int mt = 0; mt < 2; ++mt) {
#pragma unroll
        for (int nc = 0; nc < 4; ++nc) {
            const float* av = (const float*)&acc[mt][nc];
#pragma unroll
            for (int r = 0; r < 8; ++r) {
                int m  = r + 8 * lh;
                int gr = rowBase + mt * 64 + wave * 16 + m;
                int c  = colBase + nc * 16 + lm;
                size_t idx = (size_t)gr * EMBED + c;
                Y[idx] = av[r] + bias[c] + resid[idx];
            }
        }
    }
}

// ---------------------------------------------------------------------------
// LayerNorm over C=256, one wave per row, float4 vectorized.
// ---------------------------------------------------------------------------
__global__ __launch_bounds__(256, 1)
void ln_kernel(const float* __restrict__ Y, const float* __restrict__ gamma,
               const float* __restrict__ beta, float* __restrict__ out)
{
    const int row  = blockIdx.x * 8 + (threadIdx.x >> 5);
    const int lane = threadIdx.x & 31;
    const float4* yr = (const float4*)(Y + (size_t)row * EMBED);

    float4 a = yr[lane];
    float4 b = yr[lane + 32];
    float s = a.x + a.y + a.z + a.w + b.x + b.y + b.z + b.w;
#pragma unroll
    for (int off = 1; off < 32; off <<= 1) s += __shfl_xor(s, off, 32);
    const float mean = s * (1.0f / EMBED);

    float var = (a.x-mean)*(a.x-mean) + (a.y-mean)*(a.y-mean) + (a.z-mean)*(a.z-mean) + (a.w-mean)*(a.w-mean)
              + (b.x-mean)*(b.x-mean) + (b.y-mean)*(b.y-mean) + (b.z-mean)*(b.z-mean) + (b.w-mean)*(b.w-mean);
#pragma unroll
    for (int off = 1; off < 32; off <<= 1) var += __shfl_xor(var, off, 32);
    const float rstd = rsqrtf(var * (1.0f / EMBED) + 1e-5f);

    const float4 g0 = ((const float4*)gamma)[lane], g1 = ((const float4*)gamma)[lane + 32];
    const float4 e0 = ((const float4*)beta)[lane],  e1 = ((const float4*)beta)[lane + 32];
    float4* o4 = (float4*)(out + (size_t)row * EMBED);
    o4[lane]      = make_float4((a.x-mean)*rstd*g0.x+e0.x, (a.y-mean)*rstd*g0.y+e0.y,
                                (a.z-mean)*rstd*g0.z+e0.z, (a.w-mean)*rstd*g0.w+e0.w);
    o4[lane + 32] = make_float4((b.x-mean)*rstd*g1.x+e1.x, (b.y-mean)*rstd*g1.y+e1.y,
                                (b.z-mean)*rstd*g1.z+e1.z, (b.w-mean)*rstd*g1.w+e1.w);
}

// ---------------------------------------------------------------------------
extern "C" void kernel_launch(void* const* d_in, const int* in_sizes, int n_in,
                              void* d_out, int out_size, void* d_ws, size_t ws_size,
                              hipStream_t stream)
{
    (void)in_sizes; (void)n_in; (void)out_size; (void)ws_size;

    const float* q_in  = (const float*)d_in[0];
    const float* k_in  = (const float*)d_in[1];
    const float* Wq    = (const float*)d_in[2];
    const float* bq    = (const float*)d_in[3];
    const float* Wkv   = (const float*)d_in[4];
    const float* bkv   = (const float*)d_in[5];
    const float* Wo    = (const float*)d_in[6];
    const float* bo    = (const float*)d_in[7];
    const float* gamma = (const float*)d_in[8];
    const float* beta  = (const float*)d_in[9];
    float* out = (float*)d_out;

    char* ws = (char*)d_ws;
    f16*   Qw   = (f16*)  (ws + ((size_t)0  << 20));   // 4 MB  f16 [B,H,N,D]
    f16*   Kw   = (f16*)  (ws + ((size_t)4  << 20));   // 4 MB  f16 [B,H,N,D]
    f16*   Vt   = (f16*)  (ws + ((size_t)8  << 20));   // 4 MB  f16 [B,H,D,N] (transposed)
    float* attn = (float*)(ws + ((size_t)12 << 20));   // 8 MB  f32 [B,N,C]
    float* Yb   = (float*)(ws + ((size_t)20 << 20));   // 8 MB  f32 [B,N,C]

    dim3 blk(128);
    proj_kernel <<<dim3(MTOT / 128, 4),           blk, 0, stream>>>(q_in, Wq,  bq,  Qw, nullptr, EMBED);
    proj_kernel <<<dim3(MTOT / 128, 8),           blk, 0, stream>>>(k_in, Wkv, bkv, Kw, Vt,      2 * EMBED);
    flash_kernel<<<dim3(SEQ / 64, BATCH * NHEAD), blk, 0, stream>>>(Qw, Kw, Vt, attn);
    oproj_kernel<<<dim3(MTOT / 128, 4),           blk, 0, stream>>>(attn, Wo, bo, q_in, Yb);
    ln_kernel   <<<dim3(MTOT / 8), dim3(256),     0, stream>>>(Yb, gamma, beta, out);
}
